// StaticRecurrentLayer_22814866276392
// MI455X (gfx1250) — matrix operand order repaired and checked by reading, not compiled, not run on hardware
//
#include <hip/hip_runtime.h>
#include <hip/hip_bf16.h>
#include <math.h>

typedef __attribute__((ext_vector_type(16))) _Float16 v16h;
typedef __attribute__((ext_vector_type(8)))  float    v8f;

#define NN      8192
#define NBATCH  32
#define GMM     512
#define GSS     256
#define KCHUNKS 256          // NN / 32
#define STRIPS  512          // NN / 16
#define RING    16
#define APHALF  524288       // 64 * NN halfs per ring slot (1 MB)
#define CPS     4            // k-chunks per pipeline stage (16 KB staged)
#define NSTG    (KCHUNKS / CPS)   // 64 stages
#define STGB    (CPS * 4096)      // bytes per stage

// ---- workspace byte offsets ----
// Wp  : fp16 packed W (A-operand layout), 134217728 B  (fits MI455X 192MB L2)
// Ap  : ring of 16 packed-state slots (B-operand layout), 16 MB
// raw : gemm raw sums [strip][4 tiles][lane*8+r] f32, 2 MB
// u,s : current state, 1 MB each
static const size_t OFF_WP  = 0;
static const size_t OFF_AP  = 134217728;
static const size_t OFF_RAW = 150994944;
static const size_t OFF_U   = 153092096;
static const size_t OFF_S   = 154140672;

__device__ __forceinline__ v8f wmma16(v16h a, v16h b, v8f c) {
  // D = A(16x32 f16) * B(32x16 f16) + C(16x16 f32)
  return __builtin_amdgcn_wmma_f32_16x16x32_f16(
      /*neg_a=*/false, a, /*neg_b=*/false, b,
      /*c_mod=*/(short)0, c, /*reuse_a=*/false, /*reuse_b=*/false);
}

// Async global->LDS staging (ASYNCcnt-tracked); ldsaddr = per-lane LDS byte
// address (low 32 bits of generic shared pointer), gaddr = 64-bit global addr.
#define ASYNC_LOAD_B128(ldsaddr, gaddr)                              \
  asm volatile("global_load_async_to_lds_b128 %0, %1, off"           \
               :: "v"(ldsaddr), "v"(gaddr) : "memory")
#define WAIT_ASYNC(n)                                                 \
  asm volatile("s_wait_asynccnt %0" :: "i"(n) : "memory")

// ---------------------------------------------------------------------------
// Pack W (f32 row-major [i][j]) into WMMA A-operand layout, strip-major.
// A 16-bit 16x32 layout (CDNA5 ISA §7.12.2):
//   lane L: M = L&15; v<4: K = 2v + (L>=16 ? 8:0); v>=4: K = 2v + (L>=16 ?16:8)
// Tile (strip s, kchunk c) stored at dword ((s*256+c)*256 + L*8 + v).
// ---------------------------------------------------------------------------
__global__ void k_pack_w(const float* __restrict__ W, _Float16* __restrict__ Wp) {
  unsigned d = blockIdx.x * 256u + threadIdx.x;    // 0 .. NN*NN/2-1 dwords
  unsigned tile = d >> 8, t = d & 255u;
  unsigned L = t >> 3, v = t & 7u;
  unsigned strip = tile >> 8, c = tile & 255u;
  unsigned M = L & 15u, half = L >> 4;
  unsigned kb = (v < 4u) ? (2u * v + (half ? 8u : 0u))
                         : (2u * v + (half ? 16u : 8u));
  size_t i = (size_t)strip * 16u + M;
  size_t j = (size_t)c * 32u + kb;
  float w0 = W[i * NN + j];
  float w1 = W[i * NN + j + 1];
  size_t o = (size_t)d * 2;
  Wp[o]     = (_Float16)w0;
  Wp[o + 1] = (_Float16)w1;
}

// ---------------------------------------------------------------------------
// Pack state [u; s^2] into WMMA B-operand layout (32x16, K x batch).
// B 16-bit layout: lane L: n = L&15 (batch col), v<4: K = 2v + 8*(L>>4); v>=4: K = 2v + 8 + 8*(L>>4).
// Per kchunk c: 4 tiles tau = {u b0-15, u b16-31, s2 b0-15, s2 b16-31},
// tile dword base = (c*4+tau)*256, lane dwords contiguous at L*8+v.
// ---------------------------------------------------------------------------
__global__ void k_pack_a(const float* __restrict__ u, const float* __restrict__ s,
                         _Float16* __restrict__ Ap) {
  unsigned d = blockIdx.x * 256u + threadIdx.x;    // 0 .. 262143 dwords
  unsigned c = d >> 10, r = d & 1023u;
  unsigned tau = r >> 8, t = r & 255u;
  unsigned L = t >> 3, v = t & 7u;
  unsigned n = L & 15u, half = L >> 4;
  unsigned K = (v < 4u) ? (2u * v + 8u * half) : (2u * v + 8u + 8u * half);
  size_t j = (size_t)c * 32u + K;
  size_t b = (size_t)(tau & 1u) * 16u + n;
  const float* src = (tau < 2u) ? u : s;
  float x0 = src[b * NN + j];
  float x1 = src[b * NN + j + 1];
  if (tau >= 2u) { x0 *= x0; x1 *= x1; }
  size_t o = (size_t)d * 2;
  Ap[o]     = (_Float16)x0;
  Ap[o + 1] = (_Float16)x1;
}

// ---------------------------------------------------------------------------
// GEMM step: raw[i, b] (mean & var) = sum_j W[i,j] * {u, s^2}[b, j]
// 8 waves/block, each wave owns 2 strips; all waves share the per-chunk B
// (state) tiles.  B stream is staged through a 4-slot x 16KB LDS ring with
// GLOBAL_LOAD_ASYNC_TO_LDS_B128 (4 chunks/stage, lookahead 2 stages,
// ONE barrier per stage = 64 barriers per GEMM).  W (A operand) streams
// per-wave from L2 with plain b128 loads.
// Grid: 32 blocks x 256 threads = 256 waves = 512 strips.
// ---------------------------------------------------------------------------
__global__ __launch_bounds__(256) void k_gemm(const _Float16* __restrict__ Wp,
                                              const _Float16* __restrict__ Ap,
                                              float* __restrict__ raw) {
  __shared__ __align__(16) char smem[4 * STGB];    // 64 KB ring
  const int tid  = threadIdx.x;
  const int lane = tid & 31;
  const int wv   = (blockIdx.x << 3) | (tid >> 5); // 0..255
  const int s0   = wv * 2;
  const int s1   = s0 + 1;

  const char*    apb      = (const char*)Ap;
  const uint32_t lds_base = (uint32_t)(uintptr_t)smem;
  const uint32_t my_off   = (uint32_t)tid * 64u;   // 256 thr x 64B = 16KB stage

  const v16h* wp0 = (const v16h*)Wp + (size_t)s0 * KCHUNKS * 32 + lane;
  const v16h* wp1 = (const v16h*)Wp + (size_t)s1 * KCHUNKS * 32 + lane;

  v8f aU0a = {}, aU0b = {}, aV0a = {}, aV0b = {};
  v8f aU1a = {}, aU1b = {}, aV1a = {}, aV1b = {};

  // --- pipeline prologue: stages 0 and 1 in flight ---
#pragma unroll
  for (int p = 0; p < 2; ++p) {
    uint64_t g = (uint64_t)(uintptr_t)(apb + (size_t)p * STGB + my_off);
    uint32_t l = lds_base + (uint32_t)p * STGB + my_off;
#pragma unroll
    for (int q = 0; q < 4; ++q)
      ASYNC_LOAD_B128(l + (uint32_t)q * 16u, g + (uint64_t)q * 16u);
  }

#define STAGE_COMPUTE(st)                                                 \
  {                                                                       \
    const char* sb = &smem[((st) & 3) * STGB];                            \
    _Pragma("unroll")                                                     \
    for (int cc = 0; cc < CPS; ++cc) {                                    \
      int c = (st) * CPS + cc;                                            \
      const v16h* bb = (const v16h*)(sb + cc * 4096) + lane;              \
      v16h A0  = wp0[(size_t)c * 32];                                     \
      v16h A1  = wp1[(size_t)c * 32];                                     \
      v16h Bu0 = bb[0];                                                   \
      v16h Bu1 = bb[32];                                                  \
      v16h Bs0 = bb[64];                                                  \
      v16h Bs1 = bb[96];                                                  \
      aU0a = wmma16(A0, Bu0, aU0a);                                       \
      aU0b = wmma16(A0, Bu1, aU0b);                                       \
      aV0a = wmma16(A0, Bs0, aV0a);                                       \
      aV0b = wmma16(A0, Bs1, aV0b);                                       \
      aU1a = wmma16(A1, Bu0, aU1a);                                       \
      aU1b = wmma16(A1, Bu1, aU1b);                                       \
      aV1a = wmma16(A1, Bs0, aV1a);                                       \
      aV1b = wmma16(A1, Bs1, aV1b);                                       \
    }                                                                     \
  }

  // --- steady state: unconditional prefetch, wait, single barrier ---
  for (int st = 0; st < NSTG - 2; ++st) {
    int p = st + 2;
    uint64_t g = (uint64_t)(uintptr_t)(apb + (size_t)p * STGB + my_off);
    uint32_t l = lds_base + (uint32_t)(p & 3) * STGB + my_off;
#pragma unroll
    for (int q = 0; q < 4; ++q)
      ASYNC_LOAD_B128(l + (uint32_t)q * 16u, g + (uint64_t)q * 16u);
    WAIT_ASYNC(8);     // stages st+1, st+2 may stay in flight; stage st landed
    __syncthreads();   // all waves' portions of stage st visible; slot (st+2)&3
                       // was last read at stage st-2, before barrier st-1
    STAGE_COMPUTE(st);
  }

  // --- epilogue: drain last two stages ---
  WAIT_ASYNC(0);
  __syncthreads();
  STAGE_COMPUTE(NSTG - 2);
  __syncthreads();     // keep read/compute ordering symmetric for last slot
  STAGE_COMPUTE(NSTG - 1);

  // raw tile (strip, q): dword (strip*4+q)*256 + lane*8 + r,
  // element (M = r + 8*(lane>=16), batchcol = lane&15) per D layout.
  float* r0 = raw + (size_t)s0 * 4 * 256 + (size_t)lane * 8;
  *(v8f*)(r0 + 0 * 256) = aU0a;
  *(v8f*)(r0 + 1 * 256) = aU0b;
  *(v8f*)(r0 + 2 * 256) = aV0a;
  *(v8f*)(r0 + 3 * 256) = aV0b;
  float* r1 = raw + (size_t)s1 * 4 * 256 + (size_t)lane * 8;
  *(v8f*)(r1 + 0 * 256) = aU1a;
  *(v8f*)(r1 + 1 * 256) = aU1b;
  *(v8f*)(r1 + 2 * 256) = aV1a;
  *(v8f*)(r1 + 3 * 256) = aV1b;
}

// ---------------------------------------------------------------------------
// Element-wise step: add background + feedforward, sqrt, bilinear table
// lookups, leaky state update.  Decodes the WMMA D-tile layout from raw.
// ---------------------------------------------------------------------------
__global__ void k_update(const float* __restrict__ raw, int use_raw,
                         const float* __restrict__ ff_mean,
                         const float* __restrict__ ff_std,
                         const float* __restrict__ tab_u,
                         const float* __restrict__ tab_s,
                         float* __restrict__ u, float* __restrict__ s) {
  unsigned tid = blockIdx.x * 256u + threadIdx.x;   // 0 .. 262143
  unsigned b = tid >> 13;
  unsigned i = tid & 8191u;

  float rm = 0.0f, rv = 0.0f;
  if (use_raw) {
    unsigned strip = i >> 4, M = i & 15u;
    unsigned n = b & 15u, qh = b >> 4;
    unsigned L = ((M >> 3) << 4) | n;
    unsigned r = M & 7u;
    size_t base = (size_t)strip * 4u * 256u + (size_t)L * 8u + r;
    rm = raw[base + (size_t)qh * 256u];
    rv = raw[base + (size_t)(2u + qh) * 256u];
  }
  float cm  = rm + 1.0f + ff_mean[tid];                // BG_MEAN = 1.0
  float var = rv + 0.1f + ff_std[tid];                 // BG_VAR  = 0.1
  float cs  = sqrtf(fmaxf(var, 0.0f));

  float fm = (cm + 10.0f) * ((float)(GMM - 1) / 20.0f);
  fm = fminf(fmaxf(fm, 0.0f), (float)(GMM - 1) - 1e-6f);
  float fs = cs * ((float)(GSS - 1) / 10.0f);
  fs = fminf(fmaxf(fs, 0.0f), (float)(GSS - 1) - 1e-6f);
  int i0 = (int)fm, j0 = (int)fs;
  float wm  = fm - (float)i0;
  float wsf = fs - (float)j0;

  size_t o00 = (size_t)i0 * GSS + j0;
  float u00 = tab_u[o00],       u01 = tab_u[o00 + 1];
  float u10 = tab_u[o00 + GSS], u11 = tab_u[o00 + GSS + 1];
  float s00 = tab_s[o00],       s01 = tab_s[o00 + 1];
  float s10 = tab_s[o00 + GSS], s11 = tab_s[o00 + GSS + 1];
  float mu = (1.0f - wm) * ((1.0f - wsf) * u00 + wsf * u01)
           +         wm  * ((1.0f - wsf) * u10 + wsf * u11);
  float ms = (1.0f - wm) * ((1.0f - wsf) * s00 + wsf * s01)
           +         wm  * ((1.0f - wsf) * s10 + wsf * s11);

  u[tid] = 0.9f * u[tid] + 0.1f * mu;                  // a = DT/TAU = 0.1
  s[tid] = 0.9f * s[tid] + 0.1f * ms;
}

__global__ void k_zero(float* __restrict__ u, float* __restrict__ s) {
  unsigned t = blockIdx.x * 256u + threadIdx.x;
  u[t] = 0.0f;
  s[t] = 0.0f;
}

__global__ void k_out(const float* __restrict__ u, const float* __restrict__ s,
                      float* __restrict__ out) {
  unsigned t = blockIdx.x * 256u + threadIdx.x;
  out[t] = u[t];
  float sv = s[t];
  out[t + 262144u] = sv * sv;
}

extern "C" void kernel_launch(void* const* d_in, const int* in_sizes, int n_in,
                              void* d_out, int out_size, void* d_ws, size_t ws_size,
                              hipStream_t stream) {
  const float* ff_mean = (const float*)d_in[0];
  const float* ff_std  = (const float*)d_in[1];
  const float* W       = (const float*)d_in[2];
  const float* tab_u   = (const float*)d_in[3];
  const float* tab_s   = (const float*)d_in[4];

  char* ws = (char*)d_ws;
  _Float16* Wp = (_Float16*)(ws + OFF_WP);
  _Float16* Ap = (_Float16*)(ws + OFF_AP);
  float*    raw = (float*)(ws + OFF_RAW);
  float*    u   = (float*)(ws + OFF_U);
  float*    s   = (float*)(ws + OFF_S);
  float*    out = (float*)d_out;

  // One-time-per-call preprocessing (deterministic, no caching).
  k_pack_w<<<131072, 256, 0, stream>>>(W, Wp);       // 33.5M dwords
  k_zero<<<1024, 256, 0, stream>>>(u, s);

  // Delay line: GEMM at step k reads state after (k-11) updates; zero for k<12.
  for (int k = 0; k < 50; ++k) {
    int use_raw = (k >= 12) ? 1 : 0;
    if (use_raw) {
      const _Float16* slot = Ap + (size_t)((k - 11) & (RING - 1)) * APHALF;
      k_gemm<<<32, 256, 0, stream>>>(Wp, slot, raw);
    }
    k_update<<<1024, 256, 0, stream>>>(raw, use_raw, ff_mean, ff_std,
                                       tab_u, tab_s, u, s);
    if (k <= 37) {  // state m = k+1 is only consumed if m <= 38
      _Float16* ps = Ap + (size_t)((k + 1) & (RING - 1)) * APHALF;
      k_pack_a<<<1024, 256, 0, stream>>>(u, s, ps);
    }
  }
  k_out<<<1024, 256, 0, stream>>>(u, s, out);
}
